// RiemannianSGNNLayer_40888088658572
// MI455X (gfx1250) — matrix operand order repaired and checked by reading, not compiled, not run on hardware
//
#include <hip/hip_runtime.h>
#include <math.h>

typedef __attribute__((ext_vector_type(16))) __bf16 v16bf;
typedef __attribute__((ext_vector_type(8)))  float  v8f;

// ---------------------------------------------------------------- utilities
__device__ __forceinline__ float wave_sum32(float v) {
#pragma unroll
  for (int off = 16; off; off >>= 1) v += __shfl_xor(v, off, 32);
  return v;
}

// split fp32 -> bf16 hi + bf16 lo (bf16x3 GEMM emulation operands)
__device__ __forceinline__ void split_bf16(const float* buf, v16bf& h, v16bf& l) {
#pragma unroll
  for (int i = 0; i < 16; ++i) {
    float x = buf[i];
    __bf16 hh = (__bf16)x;
    h[i] = hh;
    l[i] = (__bf16)(x - (float)hh);
  }
}

// ---------------------------------------------------------------- small kernels
__global__ void k_fill0(float* p, int n) {
  int i = blockIdx.x * blockDim.x + threadIdx.x;
  if (i < n) p[i] = 0.0f;
}

__global__ void k_deg(const int* __restrict__ dst, float* __restrict__ deg, int E) {
  int i = blockIdx.x * blockDim.x + threadIdx.x;
  if (i < E) atomicAdd(&deg[dst[i]], 1.0f);
}

// deg -> d^{-1/2}, including the +1 self loop (reference concatenates arange)
__global__ void k_dinv(float* __restrict__ deg, int n) {
  int i = blockIdx.x * blockDim.x + threadIdx.x;
  if (i < n) deg[i] = rsqrtf(deg[i] + 1.0f);
}

__global__ void k_norme(const int* __restrict__ src, const int* __restrict__ dst,
                        const float* __restrict__ dinv, float* __restrict__ nrm, int E) {
  int i = blockIdx.x * blockDim.x + threadIdx.x;
  if (i < E) nrm[i] = dinv[src[i]] * dinv[dst[i]];
}

// one-shot: W (fp32) -> bf16 hi/lo operand arrays (L2-resident, shared by all waves)
__global__ void k_wsplit(const float* __restrict__ W, __bf16* __restrict__ Wh,
                         __bf16* __restrict__ Wl, int n) {
  int i = blockIdx.x * blockDim.x + threadIdx.x;
  if (i < n) {
    float x = W[i];
    __bf16 h = (__bf16)x;
    Wh[i] = h;
    Wl[i] = (__bf16)(x - (float)h);
  }
}

// agg[t][n] = xw[t][n] * dinv[n]^2   (self-loop term; also zero-initializes agg)
__global__ void k_selfinit(const float* __restrict__ xw, const float* __restrict__ dinv,
                           float* __restrict__ agg, int N, int T) {
  size_t id = (size_t)blockIdx.x * blockDim.x + threadIdx.x;
  size_t total = (size_t)T * N * 16;  // float4 chunks
  if (id >= total) return;
  int c = (int)(id & 15);
  size_t nt = id >> 4;
  int n = (int)(nt % (size_t)N);
  int t = (int)(nt / (size_t)N);
  float w = dinv[n]; w = w * w;
  size_t base = ((size_t)t * N + n) * 64 + (size_t)c * 4;
  float4 v = *(const float4*)(xw + base);
  float4 r = make_float4(v.x * w, v.y * w, v.z * w, v.w * w);
  *(float4*)(agg + base) = r;
}

// agg[t][dst] += xw[t][src] * norm[e]  (float4 read, 4 f32 atomics; per-t slice is
// 12.8 MB -> L2-resident on the 192 MB L2)
__global__ void k_scatter(const float* __restrict__ xw, float* __restrict__ agg,
                          const int* __restrict__ src, const int* __restrict__ dst,
                          const float* __restrict__ nrm, int N, int E, int T) {
  size_t id = (size_t)blockIdx.x * blockDim.x + threadIdx.x;
  size_t total = (size_t)T * E * 16;  // float4 chunks
  if (id >= total) return;
  int c = (int)(id & 15);
  size_t et = id >> 4;
  int e = (int)(et % (size_t)E);
  int t = (int)(et / (size_t)E);
  int s = src[e], d = dst[e];
  float w = nrm[e];
  float4 v = *(const float4*)(xw + ((size_t)t * N + s) * 64 + (size_t)c * 4);
  float* dp = agg + ((size_t)t * N + d) * 64 + (size_t)c * 4;
  atomicAdd(dp + 0, v.x * w);
  atomicAdd(dp + 1, v.y * w);
  atomicAdd(dp + 2, v.z * w);
  atomicAdd(dp + 3, v.w * w);
}

// ---------------------------------------------------------------- WMMA GEMM
// Out[t][n][k] = scale * (sum_{j<ksum} A[j][n][:]) . W[k][:]   (i.e. X @ W^T)
// One wave per (t, 16-row tile); 4 column groups of 16; K=64 in two K=32 chunks.
// bf16x3 split: acc += ah*bh + al*bh + ah*bl  (fp32 accumulate).
// B operands (W) are pre-split bf16 arrays -> loaded as raw 32B v16bf vectors.
__global__ void k_gemm(const float* __restrict__ A,
                       const __bf16* __restrict__ Wh, const __bf16* __restrict__ Wl,
                       float* __restrict__ Out, int N, long long slice,
                       int tcount, int ksum, float scale) {
  const int tiles = (N + 15) >> 4;
  const int w = (int)(((size_t)blockIdx.x * blockDim.x + threadIdx.x) >> 5);
  const int lane = threadIdx.x & 31;
  const int t = w / tiles;
  const int tile = w - t * tiles;
  if (t >= tcount) return;
  const int half = lane >> 4;   // 0: lanes 0-15, 1: lanes 16-31
  const int l15 = lane & 15;
  int rowA = tile * 16 + l15;
  if (rowA >= N) rowA = N - 1;

  v8f acc[4];
#pragma unroll
  for (int g = 0; g < 4; ++g)
#pragma unroll
    for (int r = 0; r < 8; ++r) acc[g][r] = 0.0f;

#pragma unroll
  for (int kc = 0; kc < 2; ++kc) {
    // --- A operand: lane<16 holds K = kb..kb+15 (kb = kc*32);
    //     lanes>=16 hold K = kb+16..kb+31.  Element e -> K=kb+half*16+e.
    float abuf[16];
    const size_t abase = (size_t)rowA * 64 + (size_t)(kc * 32 + half * 16);
#pragma unroll
    for (int seg = 0; seg < 2; ++seg) {
      const size_t off = abase + (size_t)seg * 8;
#pragma unroll
      for (int q = 0; q < 2; ++q) {
        float4 v;
        if (ksum == 1) {
          v = *(const float4*)(A + (size_t)t * slice + off + (size_t)q * 4);
        } else {
          v = make_float4(0.f, 0.f, 0.f, 0.f);
#pragma unroll
          for (int j = 0; j < 4; ++j) {
            float4 u = *(const float4*)(A + (size_t)j * slice + off + (size_t)q * 4);
            v.x += u.x; v.y += u.y; v.z += u.z; v.w += u.w;
          }
        }
        float* db = abuf + seg * 8 + q * 4;
        db[0] = v.x * scale; db[1] = v.y * scale;
        db[2] = v.z * scale; db[3] = v.w * scale;
      }
    }
    v16bf ah, al;
    split_bf16(abuf, ah, al);

#pragma unroll
    for (int g = 0; g < 4; ++g) {
      // --- B operand: column (output k) = g*16 + l15; K contiguous per lane.
      // 16-element offset multiples -> 32B-aligned vector loads of pre-split bf16.
      const size_t boff = (size_t)(g * 16 + l15) * 64 + (size_t)(kc * 32 + half * 16);
      v16bf bh = *(const v16bf*)(Wh + boff);
      v16bf bl = *(const v16bf*)(Wl + boff);
      acc[g] = __builtin_amdgcn_wmma_f32_16x16x32_bf16(false, ah, false, bh,
                                                       (short)0, acc[g], false, false);
      acc[g] = __builtin_amdgcn_wmma_f32_16x16x32_bf16(false, al, false, bh,
                                                       (short)0, acc[g], false, false);
      acc[g] = __builtin_amdgcn_wmma_f32_16x16x32_bf16(false, ah, false, bl,
                                                       (short)0, acc[g], false, false);
    }
  }

  // C/D layout: elem r -> M = r + half*8, N = l15 (within group g)
  float* outT = Out + (size_t)t * slice;
#pragma unroll
  for (int g = 0; g < 4; ++g)
#pragma unroll
    for (int r = 0; r < 8; ++r) {
      int row = tile * 16 + half * 8 + r;
      if (row < N) outT[(size_t)row * 64 + g * 16 + l15] = acc[g][r];
    }
}

// ---------------------------------------------------------------- IF neuron scan
// One wave per node; lane handles dims (lane, lane+32); norms via shfl reduction.
__global__ void k_neuron(const float* __restrict__ agg, const float* __restrict__ y,
                         const float* __restrict__ z0, float* __restrict__ o_out,
                         float* __restrict__ z_out, int N, int T) {
  const int wid = (int)(((size_t)blockIdx.x * blockDim.x + threadIdx.x) >> 5);
  const int lane = threadIdx.x & 31;
  if (wid >= N) return;
  const float EPS = 1e-7f;
  const size_t nb = (size_t)wid * 64;
  const float ya = y[nb + lane], yb = y[nb + lane + 32];
  float za = z0[nb + lane], zb = z0[nb + lane + 32];
  for (int t = 0; t < T; ++t) {
    // logmap0(z)
    float nsq = wave_sum32(za * za + zb * zb);
    float nn = sqrtf(nsq);
    float ncv = fmaxf(nn, EPS);
    float arg = fminf(fmaxf(ncv, EPS), 1.0f - 1e-5f);
    float sc = atanhf(arg) / ncv;
    const size_t xb = ((size_t)t * N + wid) * 64;
    float m0 = fmaf(sc, za, agg[xb + lane] + ya);
    float m1 = fmaf(sc, zb, agg[xb + lane + 32] + yb);
    float o0 = (m0 >= 1.0f) ? 1.0f : 0.0f;
    float o1 = (m1 >= 1.0f) ? 1.0f : 0.0f;
    o_out[xb + lane] = o0;
    o_out[xb + lane + 32] = o1;
    m0 -= o0; m1 -= o1;
    // expmap0(m)
    float msq = wave_sum32(m0 * m0 + m1 * m1);
    float mn = sqrtf(msq);
    float mc = fmaxf(mn, EPS);
    float tsc = tanhf(mc) / mc;
    za = tsc * m0; zb = tsc * m1;
  }
  z_out[nb + lane] = za;
  z_out[nb + lane + 32] = zb;
}

// ---------------------------------------------------------------- launch
extern "C" void kernel_launch(void* const* d_in, const int* in_sizes, int n_in,
                              void* d_out, int out_size, void* d_ws, size_t ws_size,
                              hipStream_t stream) {
  const float* s_seq = (const float*)d_in[0];  // [T,N,64]
  const float* z_seq = (const float*)d_in[1];  // [N,64]
  const float* Wm    = (const float*)d_in[2];  // [64,64]
  const int*   ei    = (const int*)d_in[3];    // [2,E]

  const int ND = in_sizes[1];
  const int N  = ND / 64;
  const int T  = in_sizes[0] / ND;
  const int E  = in_sizes[3] / 2;
  const int DD = in_sizes[2];                  // 64*64
  const long long slice = (long long)ND;

  // workspace layout (floats):
  //   xw[T*ND] | agg[T*ND] | y[ND] | dinv[N] | normE[E] | Wh(bf16)[DD] | Wl(bf16)[DD]
  float* ws   = (float*)d_ws;
  size_t off  = 0;
  float* xw   = ws + off; off += (size_t)T * ND;
  float* agg  = ws + off; off += (size_t)T * ND;
  float* yv   = ws + off; off += (size_t)ND;
  float* dinv = ws + off; off += (size_t)N;
  float* nrmE = ws + off; off += (size_t)E;
  off = (off + 7) & ~(size_t)7;                // 32B-align the bf16 operand arrays
  __bf16* Wh  = (__bf16*)(ws + off);           // DD bf16 = DD/2 floats
  __bf16* Wl  = Wh + (size_t)DD;

  float* o_out = (float*)d_out;                // [T,N,64]
  float* z_out = o_out + (size_t)T * ND;       // [N,64]

  const int B = 256;
  const int* esrc = ei;
  const int* edst = ei + E;

  k_fill0<<<(N + B - 1) / B, B, 0, stream>>>(dinv, N);
  k_wsplit<<<(DD + B - 1) / B, B, 0, stream>>>(Wm, Wh, Wl, DD);
  k_deg<<<(E + B - 1) / B, B, 0, stream>>>(edst, dinv, E);
  k_dinv<<<(N + B - 1) / B, B, 0, stream>>>(dinv, N);
  k_norme<<<(E + B - 1) / B, B, 0, stream>>>(esrc, edst, dinv, nrmE, E);

  {  // xw[t] = s_seq[t] @ W^T  (bf16x3-split WMMA, pre-split B)
    long long tiles = (N + 15) / 16;
    long long thr = (long long)T * tiles * 32;
    k_gemm<<<(unsigned)((thr + B - 1) / B), B, 0, stream>>>(
        s_seq, Wh, Wl, xw, N, slice, T, 1, 1.0f);
  }
  {  // agg init with self-loop term
    size_t total = (size_t)T * N * 16;
    k_selfinit<<<(unsigned)((total + B - 1) / B), B, 0, stream>>>(xw, dinv, agg, N, T);
  }
  {  // edge scatter (L2-resident per-t slices)
    size_t total = (size_t)T * E * 16;
    k_scatter<<<(unsigned)((total + B - 1) / B), B, 0, stream>>>(
        xw, agg, esrc, edst, nrmE, N, E, T);
  }
  {  // y = mean_t(agg) @ W^T  (mean fused into A load)
    long long tiles = (N + 15) / 16;
    long long thr = tiles * 32;
    k_gemm<<<(unsigned)((thr + B - 1) / B), B, 0, stream>>>(
        agg, Wh, Wl, yv, N, slice, 1, 4, 0.25f);
  }
  {  // Riemannian IF scan
    long long thr = (long long)N * 32;
    k_neuron<<<(unsigned)((thr + B - 1) / B), B, 0, stream>>>(
        agg, yv, z_seq, o_out, z_out, N, T);
  }
  (void)n_in; (void)out_size; (void)ws_size;
}
